// GCNADP_84980222918804
// MI455X (gfx1250) — matrix operand order repaired and checked by reading, not compiled, not run on hardware
//
#include <hip/hip_runtime.h>
#include <math.h>

typedef __attribute__((ext_vector_type(16))) _Float16 v16h;
typedef __attribute__((ext_vector_type(8)))  _Float16 v8h;
typedef __attribute__((ext_vector_type(8)))  float    v8f;

#define NN     8192
#define BB     4
#define KK     20
#define KDIM   40
#define KPAD   64
#define WAVES  8
#define CHUNKS (NN / (16 * WAVES))   // 64 column chunks per block

// tanh(2x) = 1 - 2/(exp(4x)+1): v_exp_f32 + v_rcp_f32, ~5 VALU vs ~30 for libm tanhf
__device__ __forceinline__ float fast_tanh2(float x) {
    float e = __expf(4.0f * x);                       // v_exp_f32 (handles +-inf/0)
    return 1.0f - 2.0f * __builtin_amdgcn_rcpf(e + 1.0f);  // raw v_rcp_f32
}
__device__ __forceinline__ float fast_tanh2_relu(float x) {
    float t = fast_tanh2(x);
    return t > 0.0f ? t : 0.0f;
}

// ---------------------------------------------------------------------------
// Pass 0: DE/EE = f16(tanh(2*x)), row-major [N][KPAD], K zero-padded 40 -> 64.
// 2 MB total; stays resident in the 192 MB L2 for the whole fused pass.
// ---------------------------------------------------------------------------
__global__ void __launch_bounds__(256)
prep_tanh_f16(const float* __restrict__ src, _Float16* __restrict__ dst) {
    int i = blockIdx.x * blockDim.x + threadIdx.x;   // over N*KPAD
    int n = i >> 6, k = i & 63;
    float v = (k < KDIM) ? fast_tanh2(src[n * KDIM + k]) : 0.0f;
    dst[i] = (_Float16)v;
}

// ---------------------------------------------------------------------------
// Fused pass: per 16-row stripe, stream all 512 column tiles through WMMA,
// score = relu(tanh(2*DE@EE^T)) + 0.01*noise, keep running top-20 per row,
// then sort indices + recompute kept weights + emit E/HEW. The 256 MB score
// matrix never touches HBM: total traffic ~= one read of noise (256 MB).
// ---------------------------------------------------------------------------
__global__ void __launch_bounds__(256)
fused_adj_topk(const _Float16* __restrict__ DEp, const _Float16* __restrict__ EEp,
               const float* __restrict__ noise, float* __restrict__ out) {
    __shared__ float stage[WAVES * 16 * 17];     // padded 16x16 score tiles
    __shared__ float lval[WAVES * 16 * KK];      // per-wave per-row top-K (ascending)
    __shared__ int   lidx[WAVES * 16 * KK];
    __shared__ float fval[16 * KK];              // merged per-row top-K
    __shared__ int   fidx[16 * KK];
    __shared__ float fw[16 * KK];

    const int tid  = threadIdx.x;
    const int lane = tid & 31;
    const int wave = tid >> 5;
    const int half = lane >> 4;
    const int l16  = lane & 15;
    const int tm   = blockIdx.x;                 // row tile [0,512)

    for (int i = tid; i < WAVES * 16 * KK; i += 256) { lval[i] = -1e30f; lidx[i] = 0; }
    __syncthreads();

    // A fragment (16 rows x K=64) -- loaded once, reused for all 64 chunks.
    // A k order used here (contiguous, same as B): lane<16 row M=lane, K={kb..kb+15};
    // lanes>=16 K={kb+16..kb+31}.
    const _Float16* aBase = DEp + (size_t)(tm * 16 + l16) * KPAD;
    union { v16h v; v8h h[2]; } a0, a1;
    a0.h[0] = *(const v8h*)(aBase +      half * 16);
    a0.h[1] = *(const v8h*)(aBase + 8  + half * 16);
    a1.h[0] = *(const v8h*)(aBase + 32 + half * 16);
    a1.h[1] = *(const v8h*)(aBase + 40 + half * 16);

    float* mylist_v = &lval[(wave * 16 + l16) * KK];
    int*   mylist_i = &lidx[(wave * 16 + l16) * KK];
    float* mystage  = &stage[wave * 16 * 17];
    float  curmin   = -1e30f;

    for (int ch = 0; ch < CHUNKS; ++ch) {
        const int tn = ch * WAVES + wave;
        // B k order used here (32x16, contiguous, same as A): lane<16 col N=lane, K={kb..kb+15}; lanes>=16 K={kb+16..kb+31}
        const _Float16* bBase = EEp + (size_t)(tn * 16 + l16) * KPAD;
        v16h b0 = *(const v16h*)(bBase +      half * 16);
        v16h b1 = *(const v16h*)(bBase + 32 + half * 16);

        v8f c = {};
        c = __builtin_amdgcn_wmma_f32_16x16x32_f16(false, a0.v, false, b0,
                                                   (short)0, c, false, false);
        c = __builtin_amdgcn_wmma_f32_16x16x32_f16(false, a1.v, false, b1,
                                                   (short)0, c, false, false);

        // epilogue: C/D layout lane L, VGPR r -> (M = r + 8*(L>=16), N = L&15)
        const int n = tn * 16 + l16;
#pragma unroll
        for (int r = 0; r < 8; ++r) {
            int m = r + half * 8;
            float sc = fast_tanh2_relu(c[r])
                     + 0.01f * noise[(size_t)(tm * 16 + m) * NN + n];
            mystage[m * 17 + l16] = sc;
        }
        __syncthreads();

        // lane r (<16) owns row r of this wave: insert 16 candidates
        if (lane < 16) {
            const int col0 = tn * 16;
            for (int j = 0; j < 16; ++j) {
                float v = mystage[lane * 17 + j];
                if (v > curmin) {                 // almost always false after warmup
                    int p = 0;
                    while (p < KK - 1 && mylist_v[p + 1] < v) {
                        mylist_v[p] = mylist_v[p + 1];
                        mylist_i[p] = mylist_i[p + 1];
                        ++p;
                    }
                    mylist_v[p] = v;
                    mylist_i[p] = col0 + j;
                    curmin = mylist_v[0];
                }
            }
        }
        __syncthreads();
    }

    // merge the 8 per-wave lists into the global top-20 per row
    if (tid < 16) {
        const int row = tid;
        float* fv = &fval[row * KK];
        int*   fi = &fidx[row * KK];
        for (int q = 0; q < KK; ++q) { fv[q] = -1e30f; fi[q] = 0; }
        for (int w = 0; w < WAVES; ++w) {
            const float* sv = &lval[(w * 16 + row) * KK];
            const int*   si = &lidx[(w * 16 + row) * KK];
            for (int q = 0; q < KK; ++q) {
                float v = sv[q];
                if (v > fv[0]) {
                    int p = 0;
                    while (p < KK - 1 && fv[p + 1] < v) {
                        fv[p] = fv[p + 1]; fi[p] = fi[p + 1]; ++p;
                    }
                    fv[p] = v; fi[p] = si[q];
                }
            }
        }
        // ascending index order (matches jnp.sort(idx))
        for (int i = 1; i < KK; ++i) {
            int vi = fi[i], j = i - 1;
            while (j >= 0 && fi[j] > vi) { fi[j + 1] = fi[j]; --j; }
            fi[j + 1] = vi;
        }
    }
    __syncthreads();

    // recompute kept weights w = relu(tanh(2 * DE[row] . EE[col]))  (20x40 MACs)
    for (int e = tid; e < 16 * KK; e += 256) {
        int row = e / KK;
        int col = fidx[e];
        const _Float16* da = DEp + (size_t)(tm * 16 + row) * KPAD;
        const _Float16* db = EEp + (size_t)col * KPAD;
        float acc = 0.0f;
#pragma unroll
        for (int k = 0; k < KDIM; ++k) acc += (float)da[k] * (float)db[k];
        fw[e] = fast_tanh2_relu(acc * 0.5f * 2.0f);   // = relu(tanh(2*acc))
    }
    __syncthreads();

    // emit E = [src; dst] then HEW, replicated over B=4 with node offsets
    const size_t BNK = (size_t)BB * NN * KK;
    for (int e = tid; e < 16 * KK; e += 256) {
        int row = e / KK, j = e % KK;
        int grow = tm * 16 + row;
        for (int b = 0; b < BB; ++b) {
            size_t pos = ((size_t)b * NN + grow) * KK + j;
            out[pos]           = (float)(grow + b * NN);      // E[0] = src
            out[BNK + pos]     = (float)(fidx[e] + b * NN);   // E[1] = dst
            out[2 * BNK + pos] = fw[e];                       // HEW
        }
    }
}

// ---------------------------------------------------------------------------
extern "C" void kernel_launch(void* const* d_in, const int* in_sizes, int n_in,
                              void* d_out, int out_size, void* d_ws, size_t ws_size,
                              hipStream_t stream) {
    // inputs: x (unused beyond B), nodevec1, nodevec2, noise
    const float* nodevec1 = (const float*)d_in[1];
    const float* nodevec2 = (const float*)d_in[2];
    const float* noise    = (const float*)d_in[3];
    float* out = (float*)d_out;

    // workspace: DEp [N*KPAD] f16 | EEp [N*KPAD] f16  (2 MB total)
    _Float16* DEp = (_Float16*)d_ws;
    _Float16* EEp = DEp + (size_t)NN * KPAD;

    prep_tanh_f16<<<(NN * KPAD) / 256, 256, 0, stream>>>(nodevec1, DEp);
    prep_tanh_f16<<<(NN * KPAD) / 256, 256, 0, stream>>>(nodevec2, EEp);
    fused_adj_topk<<<NN / 16, 256, 0, stream>>>(DEp, EEp, noise, out);
}
